// Block_3564822855835
// MI455X (gfx1250) — matrix operand order repaired and checked by reading, not compiled, not run on hardware
//
#include <hip/hip_runtime.h>

// ---------------------------------------------------------------------------
// MI455X (gfx1250, wave32) transformer block:
//   x + attn(rmsnorm(x,g1)) -> x2 ; x2 + mlp(rmsnorm(x2,g2))
// All GEMMs / attention matmuls use v_wmma_f32_16x16x32_bf16 (bf16 in, f32 acc).
// Attention V tiles are staged to LDS with GLOBAL_LOAD_ASYNC_TO_LDS_B128
// (ASYNCcnt-tracked CDNA5 async copy), synced with s_wait_asynccnt.
// ---------------------------------------------------------------------------

typedef __attribute__((ext_vector_type(16))) __bf16 bf16x16;
typedef __attribute__((ext_vector_type(8)))  __bf16 bf16x8;
typedef __attribute__((ext_vector_type(8)))  float  f32x8;

#define D_MODEL 1024
#define N_HEADS 16
#define D_HEAD  64
#define D_FF    4096
#define BATCH   2
#define SEQ     2048
#define MROWS   (BATCH * SEQ)   // 4096 token rows

static __device__ inline f32x8 zero8() {
    f32x8 v = {0.f, 0.f, 0.f, 0.f, 0.f, 0.f, 0.f, 0.f};
    return v;
}

static __device__ inline bf16x8 ldg8(const __bf16* p) {
    return *(const bf16x8*)p;
}

static __device__ inline bf16x16 cat8(bf16x8 a, bf16x8 b) {
    return __builtin_shufflevector(a, b, 0,1,2,3,4,5,6,7,8,9,10,11,12,13,14,15);
}

static __device__ inline f32x8 wmma_bf16(bf16x16 a, bf16x16 b, f32x8 c) {
    // D = A(16x32 bf16) * B(32x16 bf16) + C(16x16 f32)
    return __builtin_amdgcn_wmma_f32_16x16x32_bf16(
        false, a, false, b, (short)0, c, false, false);
}

// Workgroup-relative LDS byte offset of a shared-memory object (hardware adds
// the wave's LDS_BASE to the async-load VDST address itself).
static __device__ inline unsigned lds_offset(const void* p) {
    return (unsigned)(unsigned long long)
        (__attribute__((address_space(3))) const void*)p;
}

// A-fragment (16x32, row-major source, ld = K):
//   lane m (=lane&15): row m; hi=lane>>4 selects K chunks {8hi..8hi+7, 8hi+16..8hi+23}
static __device__ inline bf16x16 loadA(const __bf16* __restrict__ A, int ldk,
                                       int row, int k0, int hi) {
    const __bf16* p = A + (size_t)row * ldk + k0 + 8 * hi;
    return cat8(ldg8(p), ldg8(p + 16));
}

// B-fragment (32x16) from pre-transposed Bt[N][K]:
//   lane n (=lane&15): column n; hi selects K chunks {8hi..8hi+7, 8hi+16..8hi+23}
static __device__ inline bf16x16 loadBt(const __bf16* __restrict__ Bt, int ldk,
                                        int n, int k0, int hi) {
    const __bf16* p = Bt + (size_t)n * ldk + k0 + 8 * hi;
    return cat8(ldg8(p), ldg8(p + 16));
}

// ---------------------------------------------------------------------------
// fp32 [K][N] -> bf16 transposed [N][K]
// ---------------------------------------------------------------------------
__global__ __launch_bounds__(256) void cvt_transpose_k(
    const float* __restrict__ W, __bf16* __restrict__ Wt, int K, int N)
{
    int idx = blockIdx.x * 256 + threadIdx.x;
    if (idx >= K * N) return;
    int k = idx / N, n = idx % N;
    Wt[(size_t)n * K + k] = (__bf16)W[idx];
}

// ---------------------------------------------------------------------------
// RMSNorm row kernel: out_bf16 = x * rsqrt(mean(x^2)+eps) * g   (C = 1024)
// ---------------------------------------------------------------------------
__global__ __launch_bounds__(256) void rmsnorm_k(
    const float* __restrict__ x, const float* __restrict__ g,
    __bf16* __restrict__ out)
{
    const int row = blockIdx.x;
    const size_t base = (size_t)row * D_MODEL;
    const int tid = threadIdx.x;
    const int wave = tid >> 5, lane = tid & 31;

    float v0 = x[base + tid +   0], v1 = x[base + tid + 256];
    float v2 = x[base + tid + 512], v3 = x[base + tid + 768];
    float s = v0 * v0 + v1 * v1 + v2 * v2 + v3 * v3;
    for (int off = 16; off >= 1; off >>= 1)
        s += __shfl_xor(s, off, 32);

    __shared__ float wsum[8];
    __shared__ float rsv;
    if (lane == 0) wsum[wave] = s;
    __syncthreads();
    if (tid == 0) {
        float t = 0.f;
        for (int i = 0; i < 8; ++i) t += wsum[i];
        rsv = rsqrtf(t * (1.0f / D_MODEL) + 1e-6f);
    }
    __syncthreads();
    const float rs = rsv;
    out[base + tid +   0] = (__bf16)(v0 * rs * g[tid +   0]);
    out[base + tid + 256] = (__bf16)(v1 * rs * g[tid + 256]);
    out[base + tid + 512] = (__bf16)(v2 * rs * g[tid + 512]);
    out[base + tid + 768] = (__bf16)(v3 * rs * g[tid + 768]);
}

// ---------------------------------------------------------------------------
// GEMM: C[M,N] = A[M,K](bf16,row-major) @ Bt[N,K](bf16, pre-transposed weights)
// Block: 256 threads = 8 waves; block tile 256(M) x 64(N); wave tile 64x32.
// Weights (24 MB bf16 total) are L2-resident (192 MB), so fragments are read
// straight from global as 16 B loads in the exact WMMA lane layouts.
// Epilogues: 0: store bf16
//            1: outF = res + acc                     (fp32)
//            2: store bf16( gelu_exact(acc + bias) )
//            3: outF = res + acc + bias              (fp32)
// ---------------------------------------------------------------------------
template <int EPI>
__global__ __launch_bounds__(256) void gemm_bf16_k(
    const __bf16* __restrict__ A, const __bf16* __restrict__ Bt,
    int M, int N, int K,
    const float* __restrict__ bias, const float* __restrict__ res,
    float* __restrict__ outF, __bf16* __restrict__ outB)
{
    const int tid  = threadIdx.x;
    const int wave = tid >> 5, lane = tid & 31;
    const int hi   = lane >> 4, l16 = lane & 15;
    const int waveM = wave & 3, waveN = wave >> 2;
    const int Mbase = blockIdx.x * 256 + waveM * 64;
    const int Nbase = blockIdx.y * 64  + waveN * 32;

    f32x8 acc[4][2];
    for (int mt = 0; mt < 4; ++mt)
        for (int nt = 0; nt < 2; ++nt)
            acc[mt][nt] = zero8();

    for (int k0 = 0; k0 < K; k0 += 32) {
        // prefetch next A stripe into cache (global_prefetch_b8)
        if (k0 + 64 < K)
            __builtin_prefetch(A + (size_t)(Mbase + l16) * K + k0 + 64, 0, 0);

        bf16x16 bfrag[2];
        for (int nt = 0; nt < 2; ++nt)
            bfrag[nt] = loadBt(Bt, K, Nbase + nt * 16 + l16, k0, hi);
        for (int mt = 0; mt < 4; ++mt) {
            bf16x16 afrag = loadA(A, K, Mbase + mt * 16 + l16, k0, hi);
            for (int nt = 0; nt < 2; ++nt)
                acc[mt][nt] = wmma_bf16(afrag, bfrag[nt], acc[mt][nt]);
        }
    }

    for (int mt = 0; mt < 4; ++mt) {
        for (int nt = 0; nt < 2; ++nt) {
            for (int r = 0; r < 8; ++r) {
                const int row = Mbase + mt * 16 + r + 8 * hi;
                const int col = Nbase + nt * 16 + l16;
                const size_t o = (size_t)row * N + col;
                float v = acc[mt][nt][r];
                if (EPI == 0) {
                    outB[o] = (__bf16)v;
                } else if (EPI == 1) {
                    outF[o] = res[o] + v;
                } else if (EPI == 2) {
                    float t = v + bias[col];
                    float gl = 0.5f * t * (1.0f + erff(t * 0.70710678118654752f));
                    outB[o] = (__bf16)gl;
                } else { // EPI == 3
                    outF[o] = res[o] + v + bias[col];
                }
            }
        }
    }
}

// ---------------------------------------------------------------------------
// Flash-style causal attention.
// Q,K,V: bf16 [B*T, 1024] (head h occupies columns h*64 .. h*64+63)
// Grid: (B*H, T/64); block 128 = 4 waves; each wave owns a 16-row query tile.
// Per 32-key step: S = Q Kt (4 wmma), online softmax, O += P V (4 wmma).
// V tile (32 keys x 64 d) is staged row-major into LDS with
// global_load_async_to_lds_b128 (2 x 16B per thread, ASYNCcnt tracked).
// ---------------------------------------------------------------------------
__global__ __launch_bounds__(128) void attn_k(
    const __bf16* __restrict__ Q, const __bf16* __restrict__ Kb,
    const __bf16* __restrict__ Vb, __bf16* __restrict__ Y)
{
    __shared__ alignas(16) __bf16 sV[32 * D_HEAD];     // V tile row-major [key][d]
    __shared__ alignas(16) __bf16 sP[4][16 * 32];      // per-wave P tile [q][key]

    const int tid  = threadIdx.x;
    const int wave = tid >> 5, lane = tid & 31;
    const int hi   = lane >> 4, l16 = lane & 15;
    const int bh = blockIdx.x;
    const int b = bh >> 4, h = bh & 15;
    const int qblock = blockIdx.y * 64;
    const int qbase  = qblock + wave * 16;

    const unsigned svbase = lds_offset(&sV[0]);

    // Q fragments for this wave's 16 rows (d = 0..63 -> 2 K-steps of 32)
    bf16x16 aq[2];
    {
        const __bf16* qp = Q + (size_t)(b * SEQ + qbase + l16) * D_MODEL
                             + h * D_HEAD + 8 * hi;
        aq[0] = cat8(ldg8(qp), ldg8(qp + 16));
        aq[1] = cat8(ldg8(qp + 32), ldg8(qp + 48));
    }

    f32x8 O[4];
    for (int i = 0; i < 4; ++i) O[i] = zero8();
    float mrow[8], lrow[8];
    for (int r = 0; r < 8; ++r) { mrow[r] = -__builtin_inff(); lrow[r] = 0.f; }

    const int kend = qblock + 64;
    for (int kb = 0; kb < kend; kb += 32) {
        // ---- async-stage V tile (32 keys x 64 d) row-major into LDS ----
        {
            const int kk   = tid >> 2;          // local key 0..31
            const int dseg = (tid & 3) * 16;    // d segment
            const __bf16* vp = Vb + (size_t)(b * SEQ + kb + kk) * D_MODEL
                                  + h * D_HEAD + dseg;
            const unsigned loff = svbase + (unsigned)(kk * D_HEAD + dseg) * 2u;
            const unsigned long long ga = (unsigned long long)(uintptr_t)vp;
            // 32 B per thread: INST_OFFSET is added to both LDS and global sides
            asm volatile("global_load_async_to_lds_b128 %0, %1, off"
                         :: "v"(loff), "v"(ga) : "memory");
            asm volatile("global_load_async_to_lds_b128 %0, %1, off offset:16"
                         :: "v"(loff), "v"(ga) : "memory");
            asm volatile("s_wait_asynccnt 0" ::: "memory");
        }
        __syncthreads();

        // ---- S = Q @ K^T (scaled, causal-masked) ----
        f32x8 sacc[2];
        for (int nt = 0; nt < 2; ++nt) {
            sacc[nt] = zero8();
            const __bf16* kp = Kb + (size_t)(b * SEQ + kb + nt * 16 + l16) * D_MODEL
                                  + h * D_HEAD + 8 * hi;
            for (int s = 0; s < 2; ++s) {
                bf16x16 bk = cat8(ldg8(kp + s * 32), ldg8(kp + s * 32 + 16));
                sacc[nt] = wmma_bf16(aq[s], bk, sacc[nt]);
            }
        }
        for (int nt = 0; nt < 2; ++nt) {
            const int key = kb + nt * 16 + l16;
            for (int r = 0; r < 8; ++r) {
                const int q = qbase + r + 8 * hi;
                float v = sacc[nt][r] * 0.125f;          // 1/sqrt(64)
                sacc[nt][r] = (key <= q) ? v : -__builtin_inff();
            }
        }

        // ---- online softmax (row reductions across the 16-lane half) ----
        float tm[8];
        for (int r = 0; r < 8; ++r) tm[r] = fmaxf(sacc[0][r], sacc[1][r]);
        for (int off = 8; off >= 1; off >>= 1)
            for (int r = 0; r < 8; ++r)
                tm[r] = fmaxf(tm[r], __shfl_xor(tm[r], off, 32));

        float alpha[8];
        for (int r = 0; r < 8; ++r) {
            float mn = fmaxf(mrow[r], tm[r]);
            alpha[r] = __expf(mrow[r] - mn);
            mrow[r]  = mn;
        }
        float rsum[8];
        for (int r = 0; r < 8; ++r) {
            float p0 = __expf(sacc[0][r] - mrow[r]);
            float p1 = __expf(sacc[1][r] - mrow[r]);
            sacc[0][r] = p0; sacc[1][r] = p1;
            rsum[r] = p0 + p1;
        }
        for (int off = 8; off >= 1; off >>= 1)
            for (int r = 0; r < 8; ++r)
                rsum[r] += __shfl_xor(rsum[r], off, 32);
        for (int r = 0; r < 8; ++r)
            lrow[r] = lrow[r] * alpha[r] + rsum[r];
        for (int nt4 = 0; nt4 < 4; ++nt4)
            for (int r = 0; r < 8; ++r)
                O[nt4][r] *= alpha[r];

        // ---- P (C-layout) -> LDS -> A-fragment layout ----
        for (int nt = 0; nt < 2; ++nt)
            for (int r = 0; r < 8; ++r)
                sP[wave][(r + 8 * hi) * 32 + nt * 16 + l16] = (__bf16)sacc[nt][r];
        __syncthreads();

        bf16x16 ap;
        {
            const __bf16* p = &sP[wave][l16 * 32 + 8 * hi];
            ap = cat8(*(const bf16x8*)p, *(const bf16x8*)(p + 16));
        }
        // ---- O += P(16x32) @ V(32x64); B-frag: lane holds fixed d, 16 keys ----
        for (int nt4 = 0; nt4 < 4; ++nt4) {
            bf16x16 bv;
            #pragma unroll
            for (int e = 0; e < 16; ++e)
                bv[e] = sV[(8 * hi + e + (e & 8)) * D_HEAD + nt4 * 16 + l16];
            O[nt4] = wmma_bf16(ap, bv, O[nt4]);
        }
        __syncthreads();
    }

    // ---- finalize: O / l, write bf16 into [B*T, 1024] head-merged layout ----
    for (int nt4 = 0; nt4 < 4; ++nt4) {
        for (int r = 0; r < 8; ++r) {
            const size_t o = (size_t)(b * SEQ + qbase + r + 8 * hi) * D_MODEL
                           + h * D_HEAD + nt4 * 16 + l16;
            Y[o] = (__bf16)(O[nt4][r] / lrow[r]);
        }
    }
}

// ---------------------------------------------------------------------------
// Host-side orchestration
// ---------------------------------------------------------------------------
extern "C" void kernel_launch(void* const* d_in, const int* in_sizes, int n_in,
                              void* d_out, int out_size, void* d_ws, size_t ws_size,
                              hipStream_t stream)
{
    (void)in_sizes; (void)n_in; (void)out_size; (void)ws_size;
    const float* x  = (const float*)d_in[0];
    const float* Wq = (const float*)d_in[1];
    const float* Wk = (const float*)d_in[2];
    const float* Wv = (const float*)d_in[3];
    const float* Wo = (const float*)d_in[4];
    const float* W1 = (const float*)d_in[5];
    const float* b1 = (const float*)d_in[6];
    const float* W2 = (const float*)d_in[7];
    const float* b2 = (const float*)d_in[8];
    const float* g1 = (const float*)d_in[9];
    const float* g2 = (const float*)d_in[10];
    float* out = (float*)d_out;

    char*  ws  = (char*)d_ws;
    size_t off = 0;
    auto alloc = [&](size_t bytes) -> void* {
        void* p = ws + off;
        off += (bytes + 255) & ~(size_t)255;
        return p;
    };

    const size_t DM = D_MODEL, FF = D_FF, MR = MROWS;
    __bf16* wqT   = (__bf16*)alloc(DM * DM * 2);
    __bf16* wkT   = (__bf16*)alloc(DM * DM * 2);
    __bf16* wvT   = (__bf16*)alloc(DM * DM * 2);
    __bf16* woT   = (__bf16*)alloc(DM * DM * 2);
    __bf16* w1T   = (__bf16*)alloc(DM * FF * 2);
    __bf16* w2T   = (__bf16*)alloc(FF * DM * 2);
    __bf16* xn    = (__bf16*)alloc(MR * DM * 2);
    __bf16* Qb    = (__bf16*)alloc(MR * DM * 2);
    __bf16* Kb    = (__bf16*)alloc(MR * DM * 2);
    __bf16* Vb    = (__bf16*)alloc(MR * DM * 2);
    __bf16* attnY = (__bf16*)alloc(MR * DM * 2);
    float*  x2    = (float*) alloc(MR * DM * 4);
    __bf16* h2    = (__bf16*)alloc(MR * DM * 2);
    __bf16* hf    = (__bf16*)alloc(MR * FF * 2);

    // weights -> bf16, transposed [N][K]
    cvt_transpose_k<<<(DM * DM) / 256, 256, 0, stream>>>(Wq, wqT, D_MODEL, D_MODEL);
    cvt_transpose_k<<<(DM * DM) / 256, 256, 0, stream>>>(Wk, wkT, D_MODEL, D_MODEL);
    cvt_transpose_k<<<(DM * DM) / 256, 256, 0, stream>>>(Wv, wvT, D_MODEL, D_MODEL);
    cvt_transpose_k<<<(DM * DM) / 256, 256, 0, stream>>>(Wo, woT, D_MODEL, D_MODEL);
    cvt_transpose_k<<<(DM * FF) / 256, 256, 0, stream>>>(W1, w1T, D_MODEL, D_FF);
    cvt_transpose_k<<<(FF * DM) / 256, 256, 0, stream>>>(W2, w2T, D_FF, D_MODEL);

    // xn = rmsnorm(x, g1)
    rmsnorm_k<<<MROWS, 256, 0, stream>>>(x, g1, xn);

    // Q/K/V = xn @ W{q,k,v}
    dim3 gQ(MROWS / 256, D_MODEL / 64);
    gemm_bf16_k<0><<<gQ, 256, 0, stream>>>(xn, wqT, MROWS, D_MODEL, D_MODEL,
                                           nullptr, nullptr, nullptr, Qb);
    gemm_bf16_k<0><<<gQ, 256, 0, stream>>>(xn, wkT, MROWS, D_MODEL, D_MODEL,
                                           nullptr, nullptr, nullptr, Kb);
    gemm_bf16_k<0><<<gQ, 256, 0, stream>>>(xn, wvT, MROWS, D_MODEL, D_MODEL,
                                           nullptr, nullptr, nullptr, Vb);

    // causal flash attention
    attn_k<<<dim3(BATCH * N_HEADS, SEQ / 64), 128, 0, stream>>>(Qb, Kb, Vb, attnY);

    // x2 = x + attnY @ Wo
    gemm_bf16_k<1><<<gQ, 256, 0, stream>>>(attnY, woT, MROWS, D_MODEL, D_MODEL,
                                           nullptr, x, x2, nullptr);

    // h2 = rmsnorm(x2, g2)
    rmsnorm_k<<<MROWS, 256, 0, stream>>>(x2, g2, h2);

    // hf = gelu(h2 @ W1 + b1)
    dim3 g1d(MROWS / 256, D_FF / 64);
    gemm_bf16_k<2><<<g1d, 256, 0, stream>>>(h2, w1T, MROWS, D_FF, D_MODEL,
                                            b1, nullptr, nullptr, hf);

    // out = x2 + hf @ W2 + b2
    gemm_bf16_k<3><<<gQ, 256, 0, stream>>>(hf, w2T, MROWS, D_MODEL, D_FF,
                                           b2, x2, out, nullptr);
}
